// LatGatSemBertClassifier_80590766342684
// MI455X (gfx1250) — matrix operand order repaired and checked by reading, not compiled, not run on hardware
//
#include <hip/hip_runtime.h>
#include <hip/hip_bf16.h>

// ---------------------------------------------------------------------------
// MI455X (gfx1250) implementation. wave32, WMMA 16x16x32 bf16 with f32 accum.
// ---------------------------------------------------------------------------

typedef unsigned short u16;
typedef __attribute__((ext_vector_type(16))) __bf16 v16bf;
typedef __attribute__((ext_vector_type(8)))  float  v8f;
typedef __attribute__((ext_vector_type(4)))  unsigned int u32x4;

#define DEVI static __device__ __forceinline__

DEVI u16 f2bf(float f) {
    unsigned u = __float_as_uint(f);
    unsigned r = (u + 0x7FFFu + ((u >> 16) & 1u)) >> 16;
    return (u16)r;
}

union Frag { v16bf v; u16 u[16]; u32x4 q[2]; };

#define WMMA_BF16(a, b, c) \
    __builtin_amdgcn_wmma_f32_16x16x32_bf16(false, (a), false, (b), (short)0, (c), false, false)

#define LDFRAG(f, p) { (f).q[0] = *(const u32x4*)(p); (f).q[1] = *(const u32x4*)((p) + 16); }

// ---------------------------------------------------------------------------
// WMMA GEMM: C[M,N] (f32 and/or bf16 out) = A[M,K]_bf16 * BT^T + bias
//   A  : M x K row-major bf16
//   BT : N x K row-major bf16 (i.e. B pre-transposed) -> contiguous fragments
// One wave computes a 32x32 C block (2x2 WMMA tiles, 4 independent
// accumulators sharing 2 A + 2 B fragments). Software-pipelined K loop
// (two fragment sets, K/32 even for all call sites). Block = 8 waves.
// ---------------------------------------------------------------------------
__global__ __launch_bounds__(256) void wmma_gemm_kernel(
    const u16* __restrict__ A, int lda, long long sA,
    const u16* __restrict__ BT, int ldb, long long sB,
    float* __restrict__ Cf, u16* __restrict__ Cb, int ldc, long long sC,
    const float* __restrict__ bias, int M, int N, int K, int relu)
{
    int bat = blockIdx.y;
    int tilesN = N >> 5;
    int tiles  = (M >> 5) * tilesN;
    int tile   = blockIdx.x * 8 + (threadIdx.x >> 5);
    if (tile >= tiles) return;
    int tm = tile / tilesN, tn = tile % tilesN;
    int lane = threadIdx.x & 31;
    int half = lane >> 4;
    int l16  = lane & 15;

    const u16* Ab = A  + (size_t)bat * sA;
    const u16* Bb = BT + (size_t)bat * sB;

    const u16* pa0 = Ab + (size_t)(tm * 32 + l16)      * lda + half * 8;
    const u16* pa1 = Ab + (size_t)(tm * 32 + 16 + l16) * lda + half * 8;
    const u16* pb0 = Bb + (size_t)(tn * 32 + l16)      * ldb + half * 8;
    const u16* pb1 = Bb + (size_t)(tn * 32 + 16 + l16) * ldb + half * 8;

    v8f c00 = {}, c01 = {}, c10 = {}, c11 = {};
    Frag a0A, a1A, b0A, b1A, a0B, a1B, b0B, b1B;
    int steps = K >> 5;   // even (>= 2) for every call site

    LDFRAG(a0A, pa0); LDFRAG(a1A, pa1); LDFRAG(b0A, pb0); LDFRAG(b1A, pb1);
    pa0 += 32; pa1 += 32; pb0 += 32; pb1 += 32;
    for (int s = 0; s < steps - 2; s += 2) {
        LDFRAG(a0B, pa0); LDFRAG(a1B, pa1); LDFRAG(b0B, pb0); LDFRAG(b1B, pb1);
        pa0 += 32; pa1 += 32; pb0 += 32; pb1 += 32;
        c00 = WMMA_BF16(a0A.v, b0A.v, c00);
        c01 = WMMA_BF16(a0A.v, b1A.v, c01);
        c10 = WMMA_BF16(a1A.v, b0A.v, c10);
        c11 = WMMA_BF16(a1A.v, b1A.v, c11);
        LDFRAG(a0A, pa0); LDFRAG(a1A, pa1); LDFRAG(b0A, pb0); LDFRAG(b1A, pb1);
        pa0 += 32; pa1 += 32; pb0 += 32; pb1 += 32;
        c00 = WMMA_BF16(a0B.v, b0B.v, c00);
        c01 = WMMA_BF16(a0B.v, b1B.v, c01);
        c10 = WMMA_BF16(a1B.v, b0B.v, c10);
        c11 = WMMA_BF16(a1B.v, b1B.v, c11);
    }
    LDFRAG(a0B, pa0); LDFRAG(a1B, pa1); LDFRAG(b0B, pb0); LDFRAG(b1B, pb1);
    c00 = WMMA_BF16(a0A.v, b0A.v, c00);
    c01 = WMMA_BF16(a0A.v, b1A.v, c01);
    c10 = WMMA_BF16(a1A.v, b0A.v, c10);
    c11 = WMMA_BF16(a1A.v, b1A.v, c11);
    c00 = WMMA_BF16(a0B.v, b0B.v, c00);
    c01 = WMMA_BF16(a0B.v, b1B.v, c01);
    c10 = WMMA_BF16(a1B.v, b0B.v, c10);
    c11 = WMMA_BF16(a1B.v, b1B.v, c11);

    int col0 = tn * 32 + l16, col1 = col0 + 16;
    float bv0 = bias ? bias[col0] : 0.f;
    float bv1 = bias ? bias[col1] : 0.f;
    int m0 = tm * 32 + half * 8;
    int m1 = m0 + 16;
    size_t cbase = (size_t)bat * sC;
    #pragma unroll
    for (int v = 0; v < 8; v++) {
        float x00 = c00[v] + bv0, x01 = c01[v] + bv1;
        float x10 = c10[v] + bv0, x11 = c11[v] + bv1;
        if (relu) {
            x00 = fmaxf(x00, 0.f); x01 = fmaxf(x01, 0.f);
            x10 = fmaxf(x10, 0.f); x11 = fmaxf(x11, 0.f);
        }
        size_t r0 = cbase + (size_t)(m0 + v) * ldc;
        size_t r1 = cbase + (size_t)(m1 + v) * ldc;
        if (Cf) {
            Cf[r0 + col0] = x00; Cf[r0 + col1] = x01;
            Cf[r1 + col0] = x10; Cf[r1 + col1] = x11;
        }
        if (Cb) {
            Cb[r0 + col0] = f2bf(x00); Cb[r0 + col1] = f2bf(x01);
            Cb[r1 + col0] = f2bf(x10); Cb[r1 + col1] = f2bf(x11);
        }
    }
}

// ---------------------------------------------------------------------------
// Batched bf16 transpose: src (R x C) -> dst (C x R), blockIdx.z = batch.
// 32x32 LDS tiles, fully coalesced both sides.
// ---------------------------------------------------------------------------
__global__ __launch_bounds__(256) void transpose_bf16_kernel(
    const u16* __restrict__ src, u16* __restrict__ dst, int R, int C)
{
    __shared__ u16 tile[32][33];
    int b = blockIdx.z;
    int c0 = blockIdx.x * 32, r0 = blockIdx.y * 32;
    const u16* s = src + (size_t)b * R * C;
    u16* d = dst + (size_t)b * R * C;
    int tx = threadIdx.x & 31, ty = threadIdx.x >> 5;   // 32 x 8
    #pragma unroll
    for (int j = 0; j < 32; j += 8)
        tile[ty + j][tx] = s[(size_t)(r0 + ty + j) * C + c0 + tx];
    __syncthreads();
    #pragma unroll
    for (int j = 0; j < 32; j += 8)
        d[(size_t)(c0 + ty + j) * R + r0 + tx] = tile[tx][ty + j];
}

// ---------------------------------------------------------------------------
// Fused multi-head attention adjacency:
//   scores = q k^T / (sqrt(96)+1e-9), mask sm==0 -> -1e9, softmax over k,
//   mean over 8 heads, then (x*(1-eye)+eye)*sm_row  -> bf16 adjacency
// block = (b, q-tile of 16 rows), 8 waves; each head done cooperatively;
// each wave: one shared A fragment feeding two B tiles (2 indep accums).
// ---------------------------------------------------------------------------
__global__ __launch_bounds__(256) void attn_adj_kernel(
    const u16* __restrict__ q, const u16* __restrict__ kmat,
    const float* __restrict__ src_mask, u16* __restrict__ adj)
{
    __shared__ float srow[4096];
    __shared__ float acc[4096];
    int b = blockIdx.y, qt = blockIdx.x;
    int tid = threadIdx.x;
    for (int i = tid; i < 4096; i += 256) acc[i] = 0.f;
    __syncthreads();
    int wave = tid >> 5, lane = tid & 31, half = lane >> 4, l16 = lane & 15;
    int nt0 = wave * 2, nt1 = nt0 + 1;
    const float scale = 0.10206207f; // 1/(sqrt(96)+1e-9)

    for (int h = 0; h < 8; h++) {
        v8f c0 = {}, c1 = {};
        const u16* pa  = q    + ((size_t)(b * 256 + qt * 16 + l16))  * 768 + h * 96 + half * 8;
        const u16* pb0 = kmat + ((size_t)(b * 256 + nt0 * 16 + l16)) * 768 + h * 96 + half * 8;
        const u16* pb1 = kmat + ((size_t)(b * 256 + nt1 * 16 + l16)) * 768 + h * 96 + half * 8;
        #pragma unroll
        for (int ks = 0; ks < 3; ks++) {
            Frag fa, fb0, fb1;
            LDFRAG(fa, pa); LDFRAG(fb0, pb0); LDFRAG(fb1, pb1);
            pa += 32; pb0 += 32; pb1 += 32;
            c0 = WMMA_BF16(fa.v, fb0.v, c0);
            c1 = WMMA_BF16(fa.v, fb1.v, c1);
        }
        #pragma unroll
        for (int v = 0; v < 8; v++) {
            int m = half * 8 + v;
            int col0 = nt0 * 16 + l16, col1 = nt1 * 16 + l16;
            float sm0 = (col0 == 0) ? 1.f : src_mask[b * 256 + col0];
            float sm1 = src_mask[b * 256 + col1];
            srow[m * 256 + col0] = (sm0 == 0.f) ? -1e9f : c0[v] * scale;
            srow[m * 256 + col1] = (sm1 == 0.f) ? -1e9f : c1[v] * scale;
        }
        __syncthreads();
        // cooperative row softmax: 16 threads per row (lanes aligned to 16)
        int r = tid >> 4, c16 = tid & 15;
        float vals[16];
        float mx = -3.4e38f;
        #pragma unroll
        for (int j = 0; j < 16; j++) { vals[j] = srow[r * 256 + c16 + j * 16]; mx = fmaxf(mx, vals[j]); }
        mx = fmaxf(mx, __shfl_xor(mx, 1, 32)); mx = fmaxf(mx, __shfl_xor(mx, 2, 32));
        mx = fmaxf(mx, __shfl_xor(mx, 4, 32)); mx = fmaxf(mx, __shfl_xor(mx, 8, 32));
        float s = 0.f;
        #pragma unroll
        for (int j = 0; j < 16; j++) { float e = __expf(vals[j] - mx); vals[j] = e; s += e; }
        s += __shfl_xor(s, 1, 32); s += __shfl_xor(s, 2, 32);
        s += __shfl_xor(s, 4, 32); s += __shfl_xor(s, 8, 32);
        float inv = 0.125f / s; // head mean
        #pragma unroll
        for (int j = 0; j < 16; j++) acc[r * 256 + c16 + j * 16] += vals[j] * inv;
        __syncthreads();
    }
    for (int i = tid; i < 4096; i += 256) {
        int m = i >> 8, col = i & 255;
        int qr = qt * 16 + m;
        float smq = (qr == 0) ? 1.f : src_mask[b * 256 + qr];
        float val = (qr == col) ? smq : acc[i] * smq;
        adj[((size_t)b * 256 + qr) * 256 + col] = f2bf(val);
    }
}

// ---------------------------------------------------------------------------
// Fused latent adjacency:
//   lat = q2 k2^T/sqrt(768) + dep_adj + (1-sm_col)*(-1e4); softmax over k;
//   (x*(1-eye)+eye)*sm_row  -> bf16
// Shared A fragment across both N tiles; double-buffered 24-step K loop.
// ---------------------------------------------------------------------------
__global__ __launch_bounds__(256) void lat_adj_kernel(
    const u16* __restrict__ q2, const u16* __restrict__ k2,
    const float* __restrict__ dep_adj, const float* __restrict__ src_mask,
    u16* __restrict__ adj)
{
    __shared__ float srow[4096];
    int b = blockIdx.y, qt = blockIdx.x;
    int tid = threadIdx.x;
    int wave = tid >> 5, lane = tid & 31, half = lane >> 4, l16 = lane & 15;
    int nt0 = wave * 2, nt1 = nt0 + 1;
    const float scale = 0.03608439f; // 1/sqrt(768)

    const u16* pa  = q2 + ((size_t)(b * 256 + qt * 16 + l16))  * 768 + half * 8;
    const u16* pb0 = k2 + ((size_t)(b * 256 + nt0 * 16 + l16)) * 768 + half * 8;
    const u16* pb1 = k2 + ((size_t)(b * 256 + nt1 * 16 + l16)) * 768 + half * 8;

    v8f c0 = {}, c1 = {};
    Frag faA, fb0A, fb1A, faB, fb0B, fb1B;
    LDFRAG(faA, pa); LDFRAG(fb0A, pb0); LDFRAG(fb1A, pb1);
    pa += 32; pb0 += 32; pb1 += 32;
    for (int s = 0; s < 22; s += 2) {        // 24 k-steps total
        LDFRAG(faB, pa); LDFRAG(fb0B, pb0); LDFRAG(fb1B, pb1);
        pa += 32; pb0 += 32; pb1 += 32;
        c0 = WMMA_BF16(faA.v, fb0A.v, c0);
        c1 = WMMA_BF16(faA.v, fb1A.v, c1);
        LDFRAG(faA, pa); LDFRAG(fb0A, pb0); LDFRAG(fb1A, pb1);
        pa += 32; pb0 += 32; pb1 += 32;
        c0 = WMMA_BF16(faB.v, fb0B.v, c0);
        c1 = WMMA_BF16(faB.v, fb1B.v, c1);
    }
    LDFRAG(faB, pa); LDFRAG(fb0B, pb0); LDFRAG(fb1B, pb1);
    c0 = WMMA_BF16(faA.v, fb0A.v, c0);
    c1 = WMMA_BF16(faA.v, fb1A.v, c1);
    c0 = WMMA_BF16(faB.v, fb0B.v, c0);
    c1 = WMMA_BF16(faB.v, fb1B.v, c1);

    #pragma unroll
    for (int v = 0; v < 8; v++) {
        int m = half * 8 + v;
        int qr = qt * 16 + m;
        int col0 = nt0 * 16 + l16, col1 = nt1 * 16 + l16;
        float sm0 = (col0 == 0) ? 1.f : src_mask[b * 256 + col0];
        float sm1 = src_mask[b * 256 + col1];
        srow[m * 256 + col0] = c0[v] * scale + dep_adj[((size_t)b * 256 + qr) * 256 + col0]
                               + (1.f - sm0) * (-10000.f);
        srow[m * 256 + col1] = c1[v] * scale + dep_adj[((size_t)b * 256 + qr) * 256 + col1]
                               + (1.f - sm1) * (-10000.f);
    }
    __syncthreads();
    int r = tid >> 4, c16 = tid & 15;
    float vals[16];
    float mx = -3.4e38f;
    #pragma unroll
    for (int j = 0; j < 16; j++) { vals[j] = srow[r * 256 + c16 + j * 16]; mx = fmaxf(mx, vals[j]); }
    mx = fmaxf(mx, __shfl_xor(mx, 1, 32)); mx = fmaxf(mx, __shfl_xor(mx, 2, 32));
    mx = fmaxf(mx, __shfl_xor(mx, 4, 32)); mx = fmaxf(mx, __shfl_xor(mx, 8, 32));
    float s = 0.f;
    #pragma unroll
    for (int j = 0; j < 16; j++) { float e = __expf(vals[j] - mx); vals[j] = e; s += e; }
    s += __shfl_xor(s, 1, 32); s += __shfl_xor(s, 2, 32);
    s += __shfl_xor(s, 4, 32); s += __shfl_xor(s, 8, 32);
    float invs = 1.f / s;
    int qr = qt * 16 + r;
    float smq = (qr == 0) ? 1.f : src_mask[b * 256 + qr];
    #pragma unroll
    for (int j = 0; j < 16; j++) {
        int col = c16 + j * 16;
        float p = vals[j] * invs;
        float o = (qr == col) ? smq : p * smq;
        adj[((size_t)b * 256 + qr) * 256 + col] = f2bf(o);
    }
}

// ---------------------------------------------------------------------------
// Embedding gather + LayerNorm (ddof=1). One block per token.
// ---------------------------------------------------------------------------
__global__ __launch_bounds__(256) void embed_ln_kernel(
    const int* __restrict__ tok_idx, const int* __restrict__ seg_idx,
    const float* __restrict__ tok_emb, const float* __restrict__ seg_emb,
    const float* __restrict__ ln_a, const float* __restrict__ ln_b,
    float* __restrict__ gcn, u16* __restrict__ gcnbf, float* __restrict__ seq0)
{
    int t = blockIdx.x;                // 0..16383
    int b = t >> 8, l = t & 255;
    int tid = threadIdx.x;
    int idx = tok_idx[t], sg = seg_idx[t];
    float v[3], s = 0.f, s2 = 0.f;
    #pragma unroll
    for (int j = 0; j < 3; j++) {
        int d = j * 256 + tid;
        float x = tok_emb[(size_t)idx * 768 + d] + seg_emb[(size_t)sg * 768 + d];
        v[j] = x; s += x; s2 += x * x;
    }
    __shared__ float rs[256], rs2[256];
    rs[tid] = s; rs2[tid] = s2; __syncthreads();
    for (int o = 128; o > 0; o >>= 1) {
        if (tid < o) { rs[tid] += rs[tid + o]; rs2[tid] += rs2[tid + o]; }
        __syncthreads();
    }
    float mean = rs[0] * (1.f / 768.f);
    float var  = (rs2[0] - 768.f * mean * mean) * (1.f / 767.f);
    float inv  = 1.f / (sqrtf(fmaxf(var, 0.f)) + 1e-6f);
    #pragma unroll
    for (int j = 0; j < 3; j++) {
        int d = j * 256 + tid;
        float o = ln_a[d] * (v[j] - mean) * inv + ln_b[d];
        gcn[(size_t)t * 768 + d]   = o;
        gcnbf[(size_t)t * 768 + d] = f2bf(o);
        if (l == 0) seq0[(size_t)b * 768 + d] = v[j];
    }
}

__global__ __launch_bounds__(256) void pooled_kernel(
    const float* __restrict__ seq0, const float* __restrict__ pw,
    const float* __restrict__ pb, float* __restrict__ pooled)
{
    int b = blockIdx.x;
    int d = blockIdx.y * 256 + threadIdx.x;
    float acc = pb[d];
    for (int kk = 0; kk < 768; kk++)
        acc += seq0[b * 768 + kk] * pw[(size_t)kk * 768 + d];
    pooled[b * 768 + d] = tanhf(acc);
}

// f32 weight (K x N) -> bf16 transposed (N x K)
__global__ __launch_bounds__(256) void wt_bf16_kernel(
    const float* __restrict__ W, u16* __restrict__ WT, int K, int N)
{
    size_t n = (size_t)K * N;
    for (size_t i = (size_t)blockIdx.x * 256 + threadIdx.x; i < n;
         i += (size_t)gridDim.x * 256) {
        size_t nn = i / K, kk = i % K;
        WT[i] = f2bf(W[kk * (size_t)N + nn]);
    }
}

// dep tag score table (45 entries): relu(emb@fc1+b1)@fc2+b2
__global__ __launch_bounds__(256) void dep_tag_kernel(
    const float* __restrict__ dep_emb, const float* __restrict__ fc1w,
    const float* __restrict__ fc1b, const float* __restrict__ fc2w,
    const float* __restrict__ fc2b, float* __restrict__ tag_score)
{
    int tag = blockIdx.x, j = threadIdx.x;
    float h = fc1b[j];
    for (int i = 0; i < 300; i++) h += dep_emb[tag * 300 + i] * fc1w[i * 256 + j];
    h = fmaxf(h, 0.f);
    __shared__ float r[256];
    r[j] = h * fc2w[j]; __syncthreads();
    for (int o = 128; o > 0; o >>= 1) { if (j < o) r[j] += r[j + o]; __syncthreads(); }
    if (j == 0) tag_score[tag] = r[0] + fc2b[0];
}

__global__ __launch_bounds__(256) void dep_feat_kernel(
    const float* __restrict__ tag_score, const int* __restrict__ ori,
    const float* __restrict__ src_mask, float* __restrict__ df)
{
    int b = blockIdx.x, l = threadIdx.x;
    float sml = (l == 0) ? 1.f : src_mask[b * 256 + l];
    float v = tag_score[ori[b * 256 + l]] * sml + (1.f - sml) * (-1e30f);
    __shared__ float red[256];
    red[l] = v; __syncthreads();
    for (int o = 128; o > 0; o >>= 1) { if (l < o) red[l] = fmaxf(red[l], red[l + o]); __syncthreads(); }
    float mx = red[0]; __syncthreads();
    float e = __expf(v - mx);
    red[l] = e; __syncthreads();
    for (int o = 128; o > 0; o >>= 1) { if (l < o) red[l] += red[l + o]; __syncthreads(); }
    df[b * 256 + l] = e / red[0];
}

__global__ __launch_bounds__(256) void fill_ones_kernel(float* __restrict__ p, size_t n)
{
    size_t i = (size_t)blockIdx.x * 256 + threadIdx.x;
    if (i < n) p[i] = 1.f;
}

__global__ __launch_bounds__(256) void dep_scatter1_kernel(
    const int* __restrict__ head, const float* __restrict__ df, float* __restrict__ adj)
{
    int b = blockIdx.x, j = threadIdx.x;
    if (j < 255) {
        int fro = head[b * 256 + j] & 255;
        adj[((size_t)b * 256 + fro) * 256 + (j + 1)] = df[b * 256 + j];
    }
}

__global__ __launch_bounds__(256) void dep_scatter2_kernel(
    const int* __restrict__ head, const float* __restrict__ df, float* __restrict__ adj)
{
    int b = blockIdx.x, j = threadIdx.x;
    if (j < 255) {
        int fro = head[b * 256 + j] & 255;
        adj[((size_t)b * 256 + (j + 1)) * 256 + fro] = df[b * 256 + j];
    }
}

// GCN gating elementwise
__global__ __launch_bounds__(256) void gcn_gate1_kernel(
    const float* __restrict__ Isem, const float* __restrict__ Ilat,
    u16* __restrict__ Icom, size_t n)
{
    size_t i = (size_t)blockIdx.x * 256 + threadIdx.x;
    if (i >= n) return;
    float il = Ilat[i], is = Isem[i];
    float g = 1.f / (1.f + __expf(-il));
    float lg = 0.5f * g;
    Icom[i] = f2bf((1.f - lg) * is + lg * il);
}

__global__ __launch_bounds__(256) void gcn_gate2_kernel(
    float* __restrict__ H, u16* __restrict__ Hbf, const float* __restrict__ Hout, size_t n)
{
    size_t i = (size_t)blockIdx.x * 256 + threadIdx.x;
    if (i >= n) return;
    float h = H[i], ho = Hout[i];
    float gl = 1.f / (1.f + __expf(-h));
    float nh = gl * ho + (1.f - gl) * h;
    H[i] = nh; Hbf[i] = f2bf(nh);
}

// out[row] = X[row,:] . w   (D=768 dot)
__global__ __launch_bounds__(256) void dotD_kernel(
    const float* __restrict__ X, const float* __restrict__ w, float* __restrict__ out)
{
    int row = blockIdx.x, tid = threadIdx.x;
    float a = 0.f;
    #pragma unroll
    for (int j = 0; j < 3; j++) { int d = j * 256 + tid; a += X[(size_t)row * 768 + d] * w[d]; }
    __shared__ float red[256];
    red[tid] = a; __syncthreads();
    for (int o = 128; o > 0; o >>= 1) { if (tid < o) red[tid] += red[tid + o]; __syncthreads(); }
    if (tid == 0) out[row] = red[0];
}

__global__ __launch_bounds__(256) void root_fin_kernel(
    const float* __restrict__ r, const float* __restrict__ src,
    const float* __restrict__ aspect, float* __restrict__ tb)
{
    int b = blockIdx.x, l = threadIdx.x;
    float sml = (l == 0) ? 1.f : src[b * 256 + l];
    float v = r[b * 256 + l] * sml + (1.f - sml) * (-1e30f);
    __shared__ float red[256];
    red[l] = v; __syncthreads();
    for (int o = 128; o > 0; o >>= 1) { if (l < o) red[l] = fmaxf(red[l], red[l + o]); __syncthreads(); }
    float mx = red[0]; __syncthreads();
    float e = __expf(v - mx);
    red[l] = e; __syncthreads();
    for (int o = 128; o > 0; o >>= 1) { if (l < o) red[l] += red[l + o]; __syncthreads(); }
    float p = e / red[0]; __syncthreads();
    red[l] = p * aspect[b * 256 + l]; __syncthreads();
    for (int o = 128; o > 0; o >>= 1) { if (l < o) red[l] += red[l + o]; __syncthreads(); }
    if (l == 0) tb[b] = red[0];
}

__global__ __launch_bounds__(256) void lex_kernel(
    const float* __restrict__ e2, const float* __restrict__ src,
    const float* __restrict__ lex, float* __restrict__ part)
{
    int b = blockIdx.x, l = threadIdx.x;
    float v = e2[b * 256 + l];
    __shared__ float red[256];
    red[l] = v; __syncthreads();
    for (int o = 128; o > 0; o >>= 1) { if (l < o) red[l] = fmaxf(red[l], red[l + o]); __syncthreads(); }
    float mx = red[0]; __syncthreads();
    float e = __expf(v - mx);
    red[l] = e; __syncthreads();
    for (int o = 128; o > 0; o >>= 1) { if (l < o) red[l] += red[l + o]; __syncthreads(); }
    float p = e / red[0]; __syncthreads();
    float maskl = (l == 0) ? 0.f : src[b * 256 + l];
    float d = p * 50.f * maskl - lex[b * 256 + l];
    red[l] = d * d; __syncthreads();
    for (int o = 128; o > 0; o >>= 1) { if (l < o) red[l] += red[l + o]; __syncthreads(); }
    if (l == 0) part[b] = red[0];
}

__global__ __launch_bounds__(256) void aspw_kernel(
    const float* __restrict__ aspect, float* __restrict__ aspw)
{
    int b = blockIdx.x, l = threadIdx.x;
    __shared__ float red[256];
    red[l] = aspect[b * 256 + l]; __syncthreads();
    for (int o = 128; o > 0; o >>= 1) { if (l < o) red[l] += red[l + o]; __syncthreads(); }
    if (l == 0) aspw[b] = red[0];
}

// s[b,d] = sum_l H*sm ; sem_pool[b,d] = sum_l Isem*aspect / (aspw+1e-9)
__global__ __launch_bounds__(256) void rowsum_kernel(
    const float* __restrict__ H, const float* __restrict__ Isem,
    const float* __restrict__ src, const float* __restrict__ aspect,
    const float* __restrict__ aspw, float* __restrict__ S, float* __restrict__ semp)
{
    int b = blockIdx.x;
    int d = blockIdx.y * 256 + threadIdx.x;
    float s = 0.f, sp = 0.f;
    for (int l = 0; l < 256; l++) {
        float sml = (l == 0) ? 1.f : src[b * 256 + l];
        size_t o = ((size_t)b * 256 + l) * 768 + d;
        s  += H[o] * sml;
        sp += Isem[o] * aspect[b * 256 + l];
    }
    S[b * 768 + d]    = s;
    semp[b * 768 + d] = sp / (aspw[b] + 1e-9f);
}

// e[k] = (S . Isem[k]*sm_k)/1000 ; p = softmax ; outputs = p @ Isem
__global__ __launch_bounds__(256) void attn_out_kernel(
    const float* __restrict__ S, const float* __restrict__ Isem,
    const float* __restrict__ src, float* __restrict__ outv)
{
    int b = blockIdx.x, k = threadIdx.x;
    const float* sb = S + b * 768;
    const float* row = Isem + ((size_t)b * 256 + k) * 768;
    float dot = 0.f;
    for (int d = 0; d < 768; d++) dot += sb[d] * row[d];
    float smk = (k == 0) ? 1.f : src[b * 256 + k];
    float v = dot * smk * 0.001f;
    __shared__ float red[256];
    __shared__ float pe[256];
    red[k] = v; __syncthreads();
    for (int o = 128; o > 0; o >>= 1) { if (k < o) red[k] = fmaxf(red[k], red[k + o]); __syncthreads(); }
    float mx = red[0]; __syncthreads();
    float e = __expf(v - mx);
    red[k] = e; __syncthreads();
    for (int o = 128; o > 0; o >>= 1) { if (k < o) red[k] += red[k + o]; __syncthreads(); }
    pe[k] = e / red[0];
    __syncthreads();
    for (int d = threadIdx.x; d < 768; d += 256) {
        float a = 0.f;
        for (int k2 = 0; k2 < 256; k2++)
            a += pe[k2] * Isem[((size_t)b * 256 + k2) * 768 + d];
        outv[b * 768 + d] = a;
    }
}

// logits = [outputs, sem_pool, pooled] @ cls_w + cls_b ; block = 96 (3 waves)
__global__ __launch_bounds__(96) void logits_kernel(
    const float* __restrict__ outv, const float* __restrict__ semp,
    const float* __restrict__ pooled, const float* __restrict__ clsw,
    const float* __restrict__ clsb, float* __restrict__ out)
{
    int b = blockIdx.x;
    int p = threadIdx.x >> 5, lane = threadIdx.x & 31;
    float a = 0.f;
    for (int i = lane; i < 2304; i += 32) {
        float f = (i < 768) ? outv[b * 768 + i]
                : (i < 1536) ? semp[b * 768 + i - 768]
                : pooled[b * 768 + i - 1536];
        a += f * clsw[(size_t)i * 3 + p];
    }
    a += __shfl_xor(a, 16, 32); a += __shfl_xor(a, 8, 32);
    a += __shfl_xor(a, 4, 32);  a += __shfl_xor(a, 2, 32);
    a += __shfl_xor(a, 1, 32);
    if (lane == 0) out[b * 3 + p] = a + clsb[p];
}

__global__ __launch_bounds__(64) void scalar_loss_kernel(
    const float* __restrict__ tb, const float* __restrict__ lexp, float* __restrict__ out)
{
    __shared__ float r1[64], r2[64];
    int t = threadIdx.x;
    r1[t] = __logf(tb[t] + 1e-9f);
    r2[t] = lexp[t];
    __syncthreads();
    for (int o = 32; o > 0; o >>= 1) { if (t < o) { r1[t] += r1[t + o]; r2[t] += r2[t + o]; } __syncthreads(); }
    if (t == 0) {
        out[192] = -r1[0] * (1.f / 64.f);
        out[193] =  r2[0] * (1.f / (64.f * 256.f));
    }
}

// ---------------------------------------------------------------------------
extern "C" void kernel_launch(void* const* d_in, const int* in_sizes, int n_in,
                              void* d_out, int out_size, void* d_ws, size_t ws_size,
                              hipStream_t stream)
{
    (void)in_sizes; (void)n_in; (void)out_size; (void)ws_size;
    const int*   tok      = (const int*)d_in[0];
    const int*   seg      = (const int*)d_in[1];
    const float* src_mask = (const float*)d_in[3];
    const float* aspect   = (const float*)d_in[4];
    const float* lex      = (const float*)d_in[5];
    const int*   ori      = (const int*)d_in[6];
    const int*   headp    = (const int*)d_in[7];
    const float* tok_emb  = (const float*)d_in[8];
    const float* seg_emb  = (const float*)d_in[9];
    const float* pool_w   = (const float*)d_in[10];
    const float* pool_b   = (const float*)d_in[11];
    const float* ln_a     = (const float*)d_in[12];
    const float* ln_b     = (const float*)d_in[13];
    const float* dep_emb  = (const float*)d_in[14];
    const float* fc1_w    = (const float*)d_in[15];
    const float* fc1_b    = (const float*)d_in[16];
    const float* fc2_w    = (const float*)d_in[17];
    const float* fc2_b    = (const float*)d_in[18];
    const float* aq_w     = (const float*)d_in[19];
    const float* aq_b     = (const float*)d_in[20];
    const float* ak_w     = (const float*)d_in[21];
    const float* ak_b     = (const float*)d_in[22];
    const float* lq_w     = (const float*)d_in[23];
    const float* lq_b     = (const float*)d_in[24];
    const float* lk_w     = (const float*)d_in[25];
    const float* lk_b     = (const float*)d_in[26];
    const float* root_w   = (const float*)d_in[27];
    const float* gcn_w    = (const float*)d_in[28];
    const float* gcn_b    = (const float*)d_in[29];
    const float* fc3_w    = (const float*)d_in[30];
    const float* fc3_b    = (const float*)d_in[31];
    const float* cls_w    = (const float*)d_in[32];
    const float* cls_b    = (const float*)d_in[33];
    const float* vlin_w   = (const float*)d_in[34];
    float* out = (float*)d_out;

    const int B = 64, L = 256, D = 768;
    const size_t BLD = (size_t)B * L * D;       // 12,582,912
    const size_t BLL = (size_t)B * L * L;       //  4,194,304

    // ---- workspace carve ----
    char* base = (char*)d_ws; size_t off = 0;
    auto carve = [&](size_t bytes) -> char* {
        char* r = base + off; off += (bytes + 255) & ~(size_t)255; return r;
    };
    float* gcn    = (float*)carve(BLD * 4);   // H_l (starts as gcn_in)
    u16*   gcnbf  = (u16*)  carve(BLD * 2);   // H_l bf16
    float* seq0   = (float*)carve(B * D * 4);
    float* pooled = (float*)carve(B * D * 4);
    u16* qb  = (u16*)carve(BLD * 2);
    u16* kb  = (u16*)carve(BLD * 2);
    u16* q2b = (u16*)carve(BLD * 2);
    u16* k2b = (u16*)carve(BLD * 2);
    u16* aqT  = (u16*)carve((size_t)D * D * 2);
    u16* akT  = (u16*)carve((size_t)D * D * 2);
    u16* lqT  = (u16*)carve((size_t)D * D * 2);
    u16* lkT  = (u16*)carve((size_t)D * D * 2);
    u16* fc3T = (u16*)carve((size_t)D * D * 2);
    u16* g0T  = (u16*)carve((size_t)D * D * 2);
    u16* g1T  = (u16*)carve((size_t)D * D * 2);
    float* tag    = (float*)carve(256);
    float* df     = (float*)carve((size_t)B * L * 4);
    float* depadj = (float*)carve(BLL * 4);
    u16* adjag  = (u16*)carve(BLL * 2);
    u16* adjlat = (u16*)carve(BLL * 2);
    u16* HWb    = (u16*)carve(BLD * 2);
    u16* HWT    = (u16*)carve(BLD * 2);       // per-batch transpose of HW
    float* Isem = (float*)carve(BLD * 4);
    float* Ilat = (float*)carve(BLD * 4);     // reused as H_out
    u16* Icom   = (u16*)carve(BLD * 2);
    float* S    = (float*)carve(B * D * 4);
    float* semp = (float*)carve(B * D * 4);
    float* outv = (float*)carve(B * D * 4);
    float* rr   = (float*)carve((size_t)B * L * 4);
    float* e2   = (float*)carve((size_t)B * L * 4);
    float* aspw = (float*)carve(256 * 4);
    float* tb   = (float*)carve(256 * 4);
    float* lexp = (float*)carve(256 * 4);

    auto gemm = [&](const u16* A, int lda, long long sA,
                    const u16* BT, int ldb, long long sB,
                    float* Cf, u16* Cb, int ldc, long long sC,
                    const float* bias, int M, int N, int K, int relu, int nb) {
        int tiles = (M / 32) * (N / 32);
        dim3 g((tiles + 7) / 8, nb);
        hipLaunchKernelGGL(wmma_gemm_kernel, g, dim3(256), 0, stream,
                           A, lda, sA, BT, ldb, sB, Cf, Cb, ldc, sC,
                           bias, M, N, K, relu);
    };

    // 1) weight conversions (f32 KxN -> bf16 NxK)
    hipLaunchKernelGGL(wt_bf16_kernel, dim3(2304), dim3(256), 0, stream, aq_w, aqT, D, D);
    hipLaunchKernelGGL(wt_bf16_kernel, dim3(2304), dim3(256), 0, stream, ak_w, akT, D, D);
    hipLaunchKernelGGL(wt_bf16_kernel, dim3(2304), dim3(256), 0, stream, lq_w, lqT, D, D);
    hipLaunchKernelGGL(wt_bf16_kernel, dim3(2304), dim3(256), 0, stream, lk_w, lkT, D, D);
    hipLaunchKernelGGL(wt_bf16_kernel, dim3(2304), dim3(256), 0, stream, fc3_w, fc3T, D, D);
    hipLaunchKernelGGL(wt_bf16_kernel, dim3(2304), dim3(256), 0, stream, gcn_w, g0T, D, D);
    hipLaunchKernelGGL(wt_bf16_kernel, dim3(2304), dim3(256), 0, stream, gcn_w + (size_t)D * D, g1T, D, D);

    // 2) embeddings + layernorm, pooled
    hipLaunchKernelGGL(embed_ln_kernel, dim3(B * L), dim3(256), 0, stream,
                       tok, seg, tok_emb, seg_emb, ln_a, ln_b, gcn, gcnbf, seq0);
    hipLaunchKernelGGL(pooled_kernel, dim3(B, 3), dim3(256), 0, stream,
                       seq0, pool_w, pool_b, pooled);

    // 3) dep path
    hipLaunchKernelGGL(dep_tag_kernel, dim3(45), dim3(256), 0, stream,
                       dep_emb, fc1_w, fc1_b, fc2_w, fc2_b, tag);
    hipLaunchKernelGGL(dep_feat_kernel, dim3(B), dim3(256), 0, stream,
                       tag, ori, src_mask, df);
    hipLaunchKernelGGL(fill_ones_kernel, dim3((unsigned)(BLL / 256)), dim3(256), 0, stream,
                       depadj, BLL);
    hipLaunchKernelGGL(dep_scatter1_kernel, dim3(B), dim3(256), 0, stream, headp, df, depadj);
    hipLaunchKernelGGL(dep_scatter2_kernel, dim3(B), dim3(256), 0, stream, headp, df, depadj);

    // 4) projections q, k, q2, k2 (bf16 out)
    gemm(gcnbf, D, 0, aqT, D, 0, nullptr, qb,  D, 0, aq_b, B * L, D, D, 0, 1);
    gemm(gcnbf, D, 0, akT, D, 0, nullptr, kb,  D, 0, ak_b, B * L, D, D, 0, 1);
    gemm(gcnbf, D, 0, lqT, D, 0, nullptr, q2b, D, 0, lq_b, B * L, D, D, 0, 1);
    gemm(gcnbf, D, 0, lkT, D, 0, nullptr, k2b, D, 0, lk_b, B * L, D, D, 0, 1);

    // 5) adjacency matrices
    hipLaunchKernelGGL(attn_adj_kernel, dim3(16, B), dim3(256), 0, stream,
                       qb, kb, src_mask, adjag);
    hipLaunchKernelGGL(lat_adj_kernel, dim3(16, B), dim3(256), 0, stream,
                       q2b, k2b, depadj, src_mask, adjlat);

    // 6) root loss pieces (uses gcn_in before GCN loop overwrites it)
    hipLaunchKernelGGL(dotD_kernel, dim3(B * L), dim3(256), 0, stream, gcn, root_w, rr);
    hipLaunchKernelGGL(root_fin_kernel, dim3(B), dim3(256), 0, stream, rr, src_mask, aspect, tb);

    // 7) GCN layers
    const unsigned eblk = (unsigned)(BLD / 256);
    for (int layer = 0; layer < 2; layer++) {
        const u16* gT = (layer == 0) ? g0T : g1T;
        const float* gb = gcn_b + layer * D;
        // HW = H_l @ gcn_w (bf16), then per-batch transpose for the fast B path
        gemm(gcnbf, D, 0, gT, D, 0, nullptr, HWb, D, 0, nullptr, B * L, D, D, 0, 1);
        hipLaunchKernelGGL(transpose_bf16_kernel, dim3(D / 32, L / 32, B), dim3(256), 0, stream,
                           HWb, HWT, L, D);
        gemm(adjag,  L, (long long)L * L, HWT, L, (long long)L * D,
             Isem, nullptr, D, (long long)L * D, gb, L, D, L, 0, B);
        gemm(adjlat, L, (long long)L * L, HWT, L, (long long)L * D,
             Ilat, nullptr, D, (long long)L * D, gb, L, D, L, 0, B);
        hipLaunchKernelGGL(gcn_gate1_kernel, dim3(eblk), dim3(256), 0, stream,
                           Isem, Ilat, Icom, BLD);
        gemm(Icom, D, 0, fc3T, D, 0, Ilat, nullptr, D, 0, fc3_b, B * L, D, D, 1, 1);
        hipLaunchKernelGGL(gcn_gate2_kernel, dim3(eblk), dim3(256), 0, stream,
                           gcn, gcnbf, Ilat, BLD);
    }

    // 8) final pooling / attention-over-tokens / logits
    hipLaunchKernelGGL(aspw_kernel, dim3(B), dim3(256), 0, stream, aspect, aspw);
    hipLaunchKernelGGL(rowsum_kernel, dim3(B, 3), dim3(256), 0, stream,
                       gcn, Isem, src_mask, aspect, aspw, S, semp);
    hipLaunchKernelGGL(attn_out_kernel, dim3(B), dim3(256), 0, stream,
                       S, Isem, src_mask, outv);
    hipLaunchKernelGGL(logits_kernel, dim3(B), dim3(96), 0, stream,
                       outv, semp, pooled, cls_w, cls_b, out);

    // 9) lexicon loss + scalar losses
    hipLaunchKernelGGL(dotD_kernel, dim3(B * L), dim3(256), 0, stream, gcn, vlin_w, e2);
    hipLaunchKernelGGL(lex_kernel, dim3(B), dim3(256), 0, stream, e2, src_mask, lex, lexp);
    hipLaunchKernelGGL(scalar_loss_kernel, dim3(1), dim3(64), 0, stream, tb, lexp, out);
}
